// CodeExpander_8821862825991
// MI455X (gfx1250) — compile-verified, matrix operand order checked by reading, not run on hardware
//
#include <hip/hip_runtime.h>
#include <hip/hip_bf16.h>

// ---------------- model dimensions (fixed by the reference) ----------------
#define D      256
#define HD     32
#define NH     8
#define SEQ    1024
#define BATCH  4
#define FFDIM  1024
#define NENC   4
#define NDEC   4
#define KLO    1024

typedef __attribute__((ext_vector_type(16))) __bf16 bf16x16;
typedef __attribute__((ext_vector_type(8)))  __bf16 bf16x8;
typedef __attribute__((ext_vector_type(8)))  float  f32x8;

// hardware f32->bf16 (RNE) — lowers to v_cvt_pk_bf16_f32
__device__ __forceinline__ bf16x8 cvt8(f32x8 v) {
  return __builtin_convertvector(v, bf16x8);
}
__device__ __forceinline__ bf16x16 cat8(bf16x8 lo, bf16x8 hi) {
  return __builtin_shufflevector(lo, hi, 0,1,2,3,4,5,6,7,8,9,10,11,12,13,14,15);
}

// =====================================================================
// Generic GEMM: C(MxN) = A(MxK) @ W(KxN) [+ bias[N]] [+ res(MxN)]
// fp32 in/out, bf16 WMMA compute, f32 accumulation.
// Block: 256 threads (8 waves). Tile: 128x64, K step 32 (= WMMA K).
// Waves tiled 4(M) x 2(N); each wave owns a 32x32 output = 2 A-frags
// x 2 B-frags = 4 WMMAs per K step (each fragment reused twice).
// A staged row-major bf16, W staged TRANSPOSED bf16 so every fragment
// is two 16-byte ds_load_b128 runs per the CDNA5 16-bit layouts.
// =====================================================================
#define BM 128
#define BN 64
#define BK 32
#define LP 40   // LDS row pitch (bf16 elems): 80B, multiple of 16B

__global__ __launch_bounds__(256) void gemm_bf16_k(
    const float* __restrict__ A, const float* __restrict__ W,
    const float* __restrict__ bias, const float* __restrict__ res,
    float* __restrict__ C, int M, int N, int K) {
  __shared__ __bf16 As [BM][LP];   // [row][k]
  __shared__ __bf16 BsT[BN][LP];   // [col][k]  (transposed W tile)
  const int t    = threadIdx.x;
  const int wave = t >> 5, lane = t & 31;
  const int hf   = lane >> 4, l15 = lane & 15;
  const int bm = blockIdx.y * BM, bn = blockIdx.x * BN;
  const int wm = (wave >> 1) * 32;      // 0,32,64,96
  const int wn = (wave & 1) * 32;       // 0,32
  f32x8 acc00 = {0.f,0.f,0.f,0.f,0.f,0.f,0.f,0.f};
  f32x8 acc01 = acc00, acc10 = acc00, acc11 = acc00;

  // per-thread fill mapping (A: two 8-elem runs, W: one 8-elem run)
  const int ar = t >> 2, ac = (t & 3) * 8;     // A: 128 rows x 32 cols
  const int wr = t >> 3, wc = (t & 7) * 8;     // W: 32 rows x 64 cols

  for (int k0 = 0; k0 < K; k0 += BK) {
    f32x8 av0 = *(const f32x8*)&A[(size_t)(bm + ar)      * K + (k0 + ac)];
    f32x8 av1 = *(const f32x8*)&A[(size_t)(bm + ar + 64) * K + (k0 + ac)];
    f32x8 wv  = *(const f32x8*)&W[(size_t)(k0 + wr) * N + (bn + wc)];
    *(bf16x8*)&As[ar][ac]      = cvt8(av0);    // ds_store_b128
    *(bf16x8*)&As[ar + 64][ac] = cvt8(av1);
    bf16x8 wb = cvt8(wv);
#pragma unroll
    for (int j = 0; j < 8; ++j) BsT[wc + j][wr] = wb[j];   // transpose-at-store
    if (k0 + BK < K) {  // hint next K tile into cache (global_prefetch_b8)
      __builtin_prefetch(&A[(size_t)(bm + ar) * K + (k0 + BK + ac)], 0, 1);
      __builtin_prefetch(&A[(size_t)(bm + ar + 64) * K + (k0 + BK + ac)], 0, 1);
      __builtin_prefetch(&W[(size_t)(k0 + BK + wr) * N + (bn + wc)], 0, 1);
    }
    __syncthreads();

    // CDNA5 16-bit fragment: lane's K values = [kb, kb+8) U [kb+16, kb+24),
    // kb = 8*(lane>=16)  ->  two b128 LDS loads + register concat.
    const int kb = hf * 8;
    bf16x16 a0 = cat8(*(const bf16x8*)&As [wm + l15][kb],
                      *(const bf16x8*)&As [wm + l15][kb + 16]);
    bf16x16 a1 = cat8(*(const bf16x8*)&As [wm + 16 + l15][kb],
                      *(const bf16x8*)&As [wm + 16 + l15][kb + 16]);
    bf16x16 b0 = cat8(*(const bf16x8*)&BsT[wn + l15][kb],
                      *(const bf16x8*)&BsT[wn + l15][kb + 16]);
    bf16x16 b1 = cat8(*(const bf16x8*)&BsT[wn + 16 + l15][kb],
                      *(const bf16x8*)&BsT[wn + 16 + l15][kb + 16]);
    acc00 = __builtin_amdgcn_wmma_f32_16x16x32_bf16(false, a0, false, b0, (short)0, acc00, false, false);
    acc01 = __builtin_amdgcn_wmma_f32_16x16x32_bf16(false, a0, false, b1, (short)0, acc01, false, false);
    acc10 = __builtin_amdgcn_wmma_f32_16x16x32_bf16(false, a1, false, b0, (short)0, acc10, false, false);
    acc11 = __builtin_amdgcn_wmma_f32_16x16x32_bf16(false, a1, false, b1, (short)0, acc11, false, false);
    __syncthreads();
  }

  // C layout: lanes0-15 -> M=j, N=lane; lanes16-31 -> M=j+8, N=lane-16
#pragma unroll
  for (int j = 0; j < 8; ++j) {
    int row0 = bm + wm + j + hf * 8;
    int row1 = row0 + 16;
    int c0   = bn + wn + l15, c1 = c0 + 16;
    float v00 = acc00[j], v01 = acc01[j], v10 = acc10[j], v11 = acc11[j];
    if (bias) {
      float bc0 = bias[c0], bc1 = bias[c1];
      v00 += bc0; v01 += bc1; v10 += bc0; v11 += bc1;
    }
    if (res) {
      v00 += res[(size_t)row0 * N + c0]; v01 += res[(size_t)row0 * N + c1];
      v10 += res[(size_t)row1 * N + c0]; v11 += res[(size_t)row1 * N + c1];
    }
    C[(size_t)row0 * N + c0] = v00;
    C[(size_t)row0 * N + c1] = v01;
    C[(size_t)row1 * N + c0] = v10;
    C[(size_t)row1 * N + c1] = v11;
  }
}

// =====================================================================
// Flash attention (online softmax), one 16-row query tile per wave.
// Q/K/V/O are (B, L, D) with head h at column offset h*HD. HD=32 so
// q x k^T is a single bf16 WMMA (K=32). Keys processed 32 at a time.
// V is loaded one key-row per lane (contiguous b128s) and transposed
// through LDS; P round-trips LDS for the C->A fragment layout change.
// =====================================================================
__global__ __launch_bounds__(256) void flash_attn_k(
    const float* __restrict__ Qp, const float* __restrict__ Kp,
    const float* __restrict__ Vp, float* __restrict__ Op,
    int Lq, int Lk, int causal) {
  __shared__ __bf16 Plds [8][16][LP];  // per wave: [qrow][key]
  __shared__ __bf16 VldsT[8][32][LP];  // per wave: [hd][key]
  const int wave = threadIdx.x >> 5, lane = threadIdx.x & 31;
  const int hf = lane >> 4, l15 = lane & 15;
  const int b = blockIdx.z, h = blockIdx.y;
  const int qbase = blockIdx.x * 128 + wave * 16;
  const int kb = hf * 8;
  const float scale = 0.17677669529663687f;   // 1/sqrt(32)

  // q as A fragment (rows qbase..qbase+15, K = head dim), pre-scaled
  bf16x16 qa;
  {
    const float* qp = Qp + ((size_t)b * Lq + (qbase + l15)) * D + h * HD;
    f32x8 q0 = *(const f32x8*)(qp + kb);
    f32x8 q1 = *(const f32x8*)(qp + kb + 16);
    qa = cat8(cvt8(q0 * scale), cvt8(q1 * scale));
  }
  f32x8 o0 = {0.f,0.f,0.f,0.f,0.f,0.f,0.f,0.f};
  f32x8 o1 = {0.f,0.f,0.f,0.f,0.f,0.f,0.f,0.f};
  float mprev[8], lrun[8];
#pragma unroll
  for (int j = 0; j < 8; ++j) { mprev[j] = -1e30f; lrun[j] = 0.f; }

  const int nkb = causal ? ((qbase + 15) / 32 + 1) : (Lk / 32);
  for (int kbi = 0; kbi < nkb; ++kbi) {
    // ---- K^T as two B fragments (32 keys, HD=32 contraction) ----
    const float* kp0 = Kp + ((size_t)b * Lk + (kbi * 32 + l15)) * D + h * HD;
    const float* kp1 = kp0 + (size_t)16 * D;
    bf16x16 kf0 = cat8(cvt8(*(const f32x8*)(kp0 + kb)),
                       cvt8(*(const f32x8*)(kp0 + kb + 16)));
    bf16x16 kf1 = cat8(cvt8(*(const f32x8*)(kp1 + kb)),
                       cvt8(*(const f32x8*)(kp1 + kb + 16)));

    // ---- V block: lane = one key row, 32 contiguous floats; transpose to LDS
    {
      const float* vrow = Vp + ((size_t)b * Lk + (size_t)kbi * 32 + lane) * D + h * HD;
      bf16x8 vcv[4];
      vcv[0] = cvt8(*(const f32x8*)(vrow));
      vcv[1] = cvt8(*(const f32x8*)(vrow + 8));
      vcv[2] = cvt8(*(const f32x8*)(vrow + 16));
      vcv[3] = cvt8(*(const f32x8*)(vrow + 24));
#pragma unroll
      for (int d = 0; d < 32; ++d) VldsT[wave][d][lane] = vcv[d >> 3][d & 7];
    }

    f32x8 z = {0.f,0.f,0.f,0.f,0.f,0.f,0.f,0.f};
    f32x8 s0 = __builtin_amdgcn_wmma_f32_16x16x32_bf16(false, qa, false, kf0, (short)0, z, false, false);
    f32x8 s1 = __builtin_amdgcn_wmma_f32_16x16x32_bf16(false, qa, false, kf1, (short)0, z, false, false);

    // ---- online softmax; reductions stay inside each 16-lane half ----
#pragma unroll
    for (int j = 0; j < 8; ++j) {
      int r = qbase + j + hf * 8;
      float v0 = s0[j], v1 = s1[j];
      if (causal) {
        int c0 = kbi * 32 + l15;
        if (c0 > r)      v0 = -1e30f;
        if (c0 + 16 > r) v1 = -1e30f;
      }
      float mx = fmaxf(v0, v1);
      for (int o = 8; o >= 1; o >>= 1) mx = fmaxf(mx, __shfl_xor(mx, o, 32));
      float nm    = fmaxf(mprev[j], mx);
      float alpha = __expf(mprev[j] - nm);
      float p0 = __expf(v0 - nm), p1 = __expf(v1 - nm);
      float rs = p0 + p1;
      for (int o = 8; o >= 1; o >>= 1) rs += __shfl_xor(rs, o, 32);
      lrun[j]  = lrun[j] * alpha + rs;
      mprev[j] = nm;
      o0[j] *= alpha; o1[j] *= alpha;
      Plds[wave][j + hf * 8][l15]      = (__bf16)p0;   // C-layout -> LDS
      Plds[wave][j + hf * 8][16 + l15] = (__bf16)p1;
    }

    // ---- P (A frag) and V (B frags) via vector LDS loads; same-wave
    //      LDS ops are in-order on CDNA5 so no barrier is needed. ----
    bf16x16 pa  = cat8(*(const bf16x8*)&Plds [wave][l15][kb],
                       *(const bf16x8*)&Plds [wave][l15][kb + 16]);
    bf16x16 vb0 = cat8(*(const bf16x8*)&VldsT[wave][l15][kb],
                       *(const bf16x8*)&VldsT[wave][l15][kb + 16]);
    bf16x16 vb1 = cat8(*(const bf16x8*)&VldsT[wave][16 + l15][kb],
                       *(const bf16x8*)&VldsT[wave][16 + l15][kb + 16]);
    o0 = __builtin_amdgcn_wmma_f32_16x16x32_bf16(false, pa, false, vb0, (short)0, o0, false, false);
    o1 = __builtin_amdgcn_wmma_f32_16x16x32_bf16(false, pa, false, vb1, (short)0, o1, false, false);
  }

#pragma unroll
  for (int j = 0; j < 8; ++j) {
    int r = qbase + j + hf * 8;
    float inv = 1.0f / lrun[j];
    float* op = Op + ((size_t)b * Lq + r) * D + h * HD;
    op[l15]      = o0[j] * inv;
    op[16 + l15] = o1[j] * inv;
  }
}

// ---------------- elementwise / small kernels ----------------
__global__ __launch_bounds__(256) void rmsnorm_k(
    const float* __restrict__ X, const float* __restrict__ W,
    float* __restrict__ Y, int rows) {
  int wave = threadIdx.x >> 5, lane = threadIdx.x & 31;
  int row = blockIdx.x * 8 + wave;
  if (row >= rows) return;
  const float* x = X + (size_t)row * D;
  float v[8]; float ss = 0.f;
#pragma unroll
  for (int i = 0; i < 8; ++i) { v[i] = x[lane + i * 32]; ss += v[i] * v[i]; }
  for (int o = 16; o >= 1; o >>= 1) ss += __shfl_xor(ss, o, 32);
  float r = rsqrtf(ss * (1.0f / D) + 1e-6f);
  float* y = Y + (size_t)row * D;
#pragma unroll
  for (int i = 0; i < 8; ++i) y[lane + i * 32] = v[i] * r * W[lane + i * 32];
}

__global__ void rope_k(float* __restrict__ X, int total, int L) {
  int i = blockIdx.x * 256 + threadIdx.x;
  if (i >= total) return;
  int d2 = i & 15;                 // frequency index (half = 16)
  int h  = (i >> 4) & (NH - 1);
  int bl = i >> 7;                 // b*L + l   (NH*16 == 128)
  int pos = bl % L;
  float inv = __powf(10000.0f, -(float)d2 / 16.0f);
  float s, c;
  __sincosf((float)pos * inv, &s, &c);
  float* p = X + (size_t)bl * D + h * HD + d2;
  float x1 = p[0], x2 = p[16];
  p[0]  = x1 * c - x2 * s;
  p[16] = x2 * c + x1 * s;
}

__global__ void embed_k(const int* __restrict__ ids, const float* __restrict__ emb,
                        float* __restrict__ X, int total) {
  int i = blockIdx.x * 256 + threadIdx.x;
  if (i >= total) return;
  int row = i >> 8, col = i & (D - 1);
  X[i] = emb[(size_t)ids[row] * D + col];
}

__global__ void embed_shift_k(const int* __restrict__ ids, const float* __restrict__ emb,
                              float* __restrict__ X, int total, int L) {
  int i = blockIdx.x * 256 + threadIdx.x;
  if (i >= total) return;
  int col = i & (D - 1);
  int t = i >> 8;
  int l = t % L, bb = t / L;
  int id = (l == 0) ? 1 : ids[bb * L + l - 1];   // EOS = 1, shift right
  X[i] = emb[(size_t)id * D + col];
}

__global__ void swiglu_k(const float* __restrict__ G, const float* __restrict__ U,
                         float* __restrict__ Hh, int total) {
  int i = blockIdx.x * 256 + threadIdx.x;
  if (i >= total) return;
  float g = G[i];
  Hh[i] = (g / (1.0f + __expf(-g))) * U[i];
}

// =====================================================================
extern "C" void kernel_launch(void* const* d_in, const int* in_sizes, int n_in,
                              void* d_out, int out_size, void* d_ws, size_t ws_size,
                              hipStream_t stream) {
  (void)in_sizes; (void)n_in; (void)out_size; (void)ws_size;
  const int*   codes_hi = (const int*)  d_in[0];
  const int*   codes_lo = (const int*)  d_in[1];
  const float* emb_hi   = (const float*)d_in[2];
  const float* emb_lo   = (const float*)d_in[3];
  const float* enc_norm_w       = (const float*)d_in[4];
  const float* enc_attn_w       = (const float*)d_in[5];
  const float* enc_attn_b       = (const float*)d_in[6];
  const float* enc_ffn_w1       = (const float*)d_in[7];
  const float* enc_ffn_w3       = (const float*)d_in[8];
  const float* enc_ffn_w2       = (const float*)d_in[9];
  const float* enc_final_norm_w = (const float*)d_in[10];
  const float* dec_norm_w       = (const float*)d_in[11];
  const float* dec_self_w       = (const float*)d_in[12];
  const float* dec_self_b       = (const float*)d_in[13];
  const float* dec_cross_w      = (const float*)d_in[14];
  const float* dec_cross_b      = (const float*)d_in[15];
  const float* dec_ffn_w1       = (const float*)d_in[16];
  const float* dec_ffn_w3       = (const float*)d_in[17];
  const float* dec_ffn_w2       = (const float*)d_in[18];
  const float* dec_final_norm_w = (const float*)d_in[19];
  const float* out_w            = (const float*)d_in[20];
  const float* out_b            = (const float*)d_in[21];

  const int M = BATCH * SEQ;   // 4096 token rows

  // workspace arena (~67 MB of fp32)
  float* ws   = (float*)d_ws;
  float* xb   = ws;                         // encoder stream   (M*D)
  float* hb   = xb   + (size_t)M * D;       // normed           (M*D)
  float* qb   = hb   + (size_t)M * D;
  float* kb   = qb   + (size_t)M * D;
  float* vb   = kb   + (size_t)M * D;
  float* ab   = vb   + (size_t)M * D;       // attention out
  float* memb = ab   + (size_t)M * D;       // encoder memory
  float* gb   = memb + (size_t)M * D;       // ffn gate         (M*FF)
  float* ub   = gb   + (size_t)M * FFDIM;   // ffn up           (M*FF)
  float* yb   = ub   + (size_t)M * FFDIM;   // decoder stream   (M*D)

  auto gemm = [&](const float* A, const float* W, const float* bias,
                  const float* res, float* C, int Mm, int Nn, int Kk) {
    gemm_bf16_k<<<dim3(Nn / BN, Mm / BM), 256, 0, stream>>>(A, W, bias, res, C, Mm, Nn, Kk);
  };
  auto rms = [&](const float* X, const float* W, float* Y) {
    rmsnorm_k<<<M / 8, 256, 0, stream>>>(X, W, Y, M);
  };
  auto rope = [&](float* X) {
    int total = M * NH * 16;
    rope_k<<<total / 256, 256, 0, stream>>>(X, total, SEQ);
  };
  auto attn = [&](const float* Q, const float* K_, const float* V_, float* O_, int causal) {
    flash_attn_k<<<dim3(SEQ / 128, NH, BATCH), 256, 0, stream>>>(Q, K_, V_, O_, SEQ, SEQ, causal);
  };

  // ---------------- encoder ----------------
  embed_k<<<(M * D) / 256, 256, 0, stream>>>(codes_hi, emb_hi, xb, M * D);
  for (int i = 0; i < NENC; ++i) {
    const float* wq = enc_attn_w + (size_t)(i * 4 + 0) * D * D;
    const float* wk = enc_attn_w + (size_t)(i * 4 + 1) * D * D;
    const float* wv = enc_attn_w + (size_t)(i * 4 + 2) * D * D;
    const float* wo = enc_attn_w + (size_t)(i * 4 + 3) * D * D;
    const float* bq = enc_attn_b + (size_t)(i * 4 + 0) * D;
    const float* bk2 = enc_attn_b + (size_t)(i * 4 + 1) * D;
    const float* bv2 = enc_attn_b + (size_t)(i * 4 + 2) * D;
    const float* bo = enc_attn_b + (size_t)(i * 4 + 3) * D;
    rms(xb, enc_norm_w + (size_t)(i * 2 + 0) * D, hb);
    gemm(hb, wq, bq, nullptr, qb, M, D, D);
    gemm(hb, wk, bk2, nullptr, kb, M, D, D);
    gemm(hb, wv, bv2, nullptr, vb, M, D, D);
    rope(qb); rope(kb);
    attn(qb, kb, vb, ab, 0);
    gemm(ab, wo, bo, xb, xb, M, D, D);                       // + residual
    rms(xb, enc_norm_w + (size_t)(i * 2 + 1) * D, hb);
    gemm(hb, enc_ffn_w1 + (size_t)i * D * FFDIM, nullptr, nullptr, gb, M, FFDIM, D);
    gemm(hb, enc_ffn_w3 + (size_t)i * D * FFDIM, nullptr, nullptr, ub, M, FFDIM, D);
    swiglu_k<<<(M * FFDIM) / 256, 256, 0, stream>>>(gb, ub, gb, M * FFDIM);
    gemm(gb, enc_ffn_w2 + (size_t)i * FFDIM * D, nullptr, xb, xb, M, D, FFDIM);
  }
  rms(xb, enc_final_norm_w, memb);

  // ---------------- decoder ----------------
  embed_shift_k<<<(M * D) / 256, 256, 0, stream>>>(codes_lo, emb_lo, yb, M * D, SEQ);
  for (int i = 0; i < NDEC; ++i) {
    // self-attention (causal)
    {
      const float* wq = dec_self_w + (size_t)(i * 4 + 0) * D * D;
      const float* wk = dec_self_w + (size_t)(i * 4 + 1) * D * D;
      const float* wv = dec_self_w + (size_t)(i * 4 + 2) * D * D;
      const float* wo = dec_self_w + (size_t)(i * 4 + 3) * D * D;
      const float* bq = dec_self_b + (size_t)(i * 4 + 0) * D;
      const float* bk2 = dec_self_b + (size_t)(i * 4 + 1) * D;
      const float* bv2 = dec_self_b + (size_t)(i * 4 + 2) * D;
      const float* bo = dec_self_b + (size_t)(i * 4 + 3) * D;
      rms(yb, dec_norm_w + (size_t)(i * 3 + 0) * D, hb);
      gemm(hb, wq, bq, nullptr, qb, M, D, D);
      gemm(hb, wk, bk2, nullptr, kb, M, D, D);
      gemm(hb, wv, bv2, nullptr, vb, M, D, D);
      rope(qb); rope(kb);
      attn(qb, kb, vb, ab, 1);
      gemm(ab, wo, bo, yb, yb, M, D, D);
    }
    // cross-attention (kv from encoder memory; rope on both, per reference)
    {
      const float* wq = dec_cross_w + (size_t)(i * 4 + 0) * D * D;
      const float* wk = dec_cross_w + (size_t)(i * 4 + 1) * D * D;
      const float* wv = dec_cross_w + (size_t)(i * 4 + 2) * D * D;
      const float* wo = dec_cross_w + (size_t)(i * 4 + 3) * D * D;
      const float* bq = dec_cross_b + (size_t)(i * 4 + 0) * D;
      const float* bk2 = dec_cross_b + (size_t)(i * 4 + 1) * D;
      const float* bv2 = dec_cross_b + (size_t)(i * 4 + 2) * D;
      const float* bo = dec_cross_b + (size_t)(i * 4 + 3) * D;
      rms(yb, dec_norm_w + (size_t)(i * 3 + 1) * D, hb);
      gemm(hb, wq, bq, nullptr, qb, M, D, D);
      gemm(memb, wk, bk2, nullptr, kb, M, D, D);
      gemm(memb, wv, bv2, nullptr, vb, M, D, D);
      rope(qb); rope(kb);
      attn(qb, kb, vb, ab, 0);
      gemm(ab, wo, bo, yb, yb, M, D, D);
    }
    // FFN
    rms(yb, dec_norm_w + (size_t)(i * 3 + 2) * D, hb);
    gemm(hb, dec_ffn_w1 + (size_t)i * D * FFDIM, nullptr, nullptr, gb, M, FFDIM, D);
    gemm(hb, dec_ffn_w3 + (size_t)i * D * FFDIM, nullptr, nullptr, ub, M, FFDIM, D);
    swiglu_k<<<(M * FFDIM) / 256, 256, 0, stream>>>(gb, ub, gb, M * FFDIM);
    gemm(gb, dec_ffn_w2 + (size_t)i * FFDIM * D, nullptr, yb, yb, M, D, FFDIM);
  }

  // ---------------- logits ----------------
  rms(yb, dec_final_norm_w, hb);
  gemm(hb, out_w, out_b, nullptr, (float*)d_out, M, KLO, D);
}
